// update_e_6975026889057
// MI455X (gfx1250) — compile-verified, matrix operand order checked by reading, not run on hardware
//
#include <hip/hip_runtime.h>
#include <stdint.h>

typedef __attribute__((ext_vector_type(16))) __bf16 v16bf;
typedef __attribute__((ext_vector_type(8)))  float  v8f;

#define DEV static __device__ __forceinline__

// ---- helpers -------------------------------------------------------------

DEV uint32_t pk_bf16(float a, float b) {     // two f32 -> packed bf16 dword
    union { __bf16 h[2]; uint32_t u; } x;
    x.h[0] = (__bf16)a; x.h[1] = (__bf16)b;  // native cvt on gfx1250
    return x.u;
}

union BPack { uint32_t u[8]; uint4 q[2]; v16bf v; };

DEV v16bf lds_read_op(const uint32_t* p) {   // 8 dwords (16 bf16), 16B-aligned
    BPack b;
    b.q[0] = *reinterpret_cast<const uint4*>(p);
    b.q[1] = *reinterpret_cast<const uint4*>(p + 4);
    return b.v;
}

DEV float ssp(float x) {                     // softplus(x) - log(2), fast
    float sp = (x > 20.f) ? x : __logf(1.f + __expf(x));
    return sp - 0.69314718055994531f;
}

// 16-bit WMMA operand layout: lane holds two 8-element K-runs at kb, kb+16,
// kb = 32*kstep + 8*(lane>=16). Dword d holds elements {2d, 2d+1} of the run.
DEV int kofs(int d) { return (d < 4) ? 2 * d : 16 + 2 * (d - 4); }

// ---- kernel 1: v_ = v@lin_w.T and v_hull = v@lin_hull_w.T ----------------

__global__ __launch_bounds__(256) void proj_kernel(
    const float* __restrict__ v, const float* __restrict__ w1,
    const float* __restrict__ w2, float* __restrict__ o1,
    float* __restrict__ o2, int N)
{
    __shared__ __align__(16) uint32_t sB[2][8 * 4 * 32 * 8];  // 64 KB
    const int tid = threadIdx.x;
    const int wave = tid >> 5, lane = tid & 31;
    const int hh = lane >> 4, l16 = lane & 15;

    for (int w = 0; w < 2; ++w) {
        const float* wp = w ? w2 : w1;
        for (int idx = tid; idx < 8 * 4 * 32 * 8; idx += 256) {
            int d = idx & 7; int r = idx >> 3;
            int L = r & 31;  r >>= 5;
            int s = r & 3;   int t = r >> 2;
            int n = 16 * t + (L & 15);
            int k = 32 * s + 8 * (L >> 4) + kofs(d);
            sB[w][idx] = pk_bf16(wp[n * 128 + k], wp[n * 128 + k + 1]);
        }
    }
    __syncthreads();

    const int rbase = blockIdx.x * 128 + wave * 16;      // strip row base
    const bool full = blockIdx.x * 128 + 128 <= N;
    int row = rbase + l16;
    bool ok = full || row < N;
    const float* vp = v + (size_t)(ok ? row : 0) * 128;
    v16bf a[4];
#pragma unroll
    for (int s = 0; s < 4; ++s) {
        int kb = 32 * s + 8 * hh;
        v16bf av;
#pragma unroll
        for (int d = 0; d < 8; ++d) {
            int k = kb + kofs(d);
            av[2 * d]     = (__bf16)(ok ? vp[k]     : 0.f);
            av[2 * d + 1] = (__bf16)(ok ? vp[k + 1] : 0.f);
        }
        a[s] = av;
    }

#pragma unroll
    for (int w = 0; w < 2; ++w) {
        float* obase = (w ? o2 : o1) + (size_t)(rbase + 8 * hh) * 128 + l16;
        for (int t = 0; t < 8; ++t) {
            v8f acc = {};
#pragma unroll
            for (int s = 0; s < 4; ++s) {
                v16bf bm = lds_read_op(&sB[w][((t * 4 + s) * 32 + lane) * 8]);
                acc = __builtin_amdgcn_wmma_f32_16x16x32_bf16(
                        false, a[s], false, bm, (short)0, acc, false, false);
            }
            if (full) {
#pragma unroll
                for (int i = 0; i < 8; ++i)
                    obase[i * 128 + t * 16] = acc[i];        // imm offsets
            } else {
#pragma unroll
                for (int i = 0; i < 8; ++i)
                    if (rbase + 8 * hh + i < N)
                        obase[i * 128 + t * 16] = acc[i];
            }
        }
    }
}

// ---- kernel 2: fused per-edge filter MLP + cutoff + gather ---------------
// out[e,:] = v_[j[e],:] * ( ssp(emb[e,:]@W1^T + b1) @ W2^T + b2 ) * C(e)

template <int KIN, int KS1, bool HAS_CUT>
__global__ __launch_bounds__(256) void edge_kernel(
    const float* __restrict__ gemb,   // [E, KIN]
    const float* __restrict__ dist,   // [E] (unused if !HAS_CUT)
    const int*   __restrict__ jidx,   // row 0 of edge_index
    const float* __restrict__ vproj,  // [N,128]
    const float* __restrict__ w1, const float* __restrict__ b1,
    const float* __restrict__ w2, const float* __restrict__ b2,
    float* __restrict__ out, int E, int numTiles)
{
    constexpr int HPAD = 136;                       // 272B rows: conflict-free
    __shared__ __align__(16) uint32_t sB1[8 * KS1 * 32 * 8];
    __shared__ __align__(16) uint32_t sB2[8 * 4 * 32 * 8];
    __shared__ __align__(16) __bf16   sH[8 * 16 * HPAD];   // per-wave h strips
    __shared__ float sb1[128], sb2[128], sC[128];
    __shared__ int   sJ[128];

    const int tid = threadIdx.x;
    const int wave = tid >> 5, lane = tid & 31;
    const int hh = lane >> 4, l16 = lane & 15;

    // Build W1/W2 B-operand images (lane-major WMMA layout) once per block.
    for (int idx = tid; idx < 8 * KS1 * 32 * 8; idx += 256) {
        int d = idx & 7; int r = idx >> 3;
        int L = r & 31;  r >>= 5;
        int s = r % KS1; int t = r / KS1;
        int n = 16 * t + (L & 15);
        int k = 32 * s + 8 * (L >> 4) + kofs(d);
        float x0 = (k     < KIN) ? w1[n * KIN + k]     : 0.f;
        float x1 = (k + 1 < KIN) ? w1[n * KIN + k + 1] : 0.f;
        sB1[idx] = pk_bf16(x0, x1);
    }
    for (int idx = tid; idx < 8 * 4 * 32 * 8; idx += 256) {
        int d = idx & 7; int r = idx >> 3;
        int L = r & 31;  r >>= 5;
        int s = r & 3;   int t = r >> 2;
        int n = 16 * t + (L & 15);
        int k = 32 * s + 8 * (L >> 4) + kofs(d);
        sB2[idx] = pk_bf16(w2[n * 128 + k], w2[n * 128 + k + 1]);
    }
    if (tid < 128) { sb1[tid] = b1[tid]; sb2[tid] = b2[tid]; }
    __syncthreads();

    __bf16* hstore = &sH[(wave * 16 + 8 * hh) * HPAD + l16];   // D-layout spill
    const __bf16* hrow = &sH[(wave * 16 + l16) * HPAD];        // A-layout read

    for (int tile = blockIdx.x; tile < numTiles; tile += gridDim.x) {
        __syncthreads();                        // tile-scoped sJ/sC reuse
        const bool full = tile * 128 + 128 <= E;
        if (tid < 128) {
            int e = tile * 128 + tid;
            bool eo = full || e < E;
            sJ[tid] = eo ? jidx[e] : 0;
            if (HAS_CUT)
                sC[tid] = eo ? 0.5f * (__cosf(dist[e] * 0.314159265358979f) + 1.f)
                             : 0.f;
        }
        __syncthreads();

        // GEMM1 A operand: gathered rows of gemb, zero-padded past KIN.
        int ge = tile * 128 + wave * 16 + l16;
        bool eok = full || ge < E;
        const float* gp = gemb + (size_t)(eok ? ge : 0) * KIN;
        v16bf a1[KS1];
#pragma unroll
        for (int s = 0; s < KS1; ++s) {
            int kb = 32 * s + 8 * hh;
            v16bf av;
#pragma unroll
            for (int d = 0; d < 8; ++d) {
                int k = kb + kofs(d);
                av[2 * d]     = (__bf16)((eok && k     < KIN) ? gp[k]     : 0.f);
                av[2 * d + 1] = (__bf16)((eok && k + 1 < KIN) ? gp[k + 1] : 0.f);
            }
            a1[s] = av;
        }
        {   // prefetch next tile's row for this lane
            int nge = (tile + (int)gridDim.x) * 128 + wave * 16 + l16;
            if (nge < E) __builtin_prefetch(gemb + (size_t)nge * KIN, 0, 0);
        }

        // GEMM1: h = ssp(A@W1^T + b1), spill bf16 strip to LDS (imm offsets).
#pragma unroll
        for (int t = 0; t < 8; ++t) {
            v8f acc = {};
#pragma unroll
            for (int s = 0; s < KS1; ++s) {
                v16bf bm = lds_read_op(&sB1[((t * KS1 + s) * 32 + lane) * 8]);
                acc = __builtin_amdgcn_wmma_f32_16x16x32_bf16(
                        false, a1[s], false, bm, (short)0, acc, false, false);
            }
            float bias = sb1[16 * t + l16];
#pragma unroll
            for (int i = 0; i < 8; ++i)
                hstore[i * HPAD + 16 * t] = (__bf16)ssp(acc[i] + bias);
        }

        // GEMM2 A operand: reload own strip in WMMA A layout (intra-wave LDS).
        v16bf a2[4];
#pragma unroll
        for (int s = 0; s < 4; ++s) {
            int kb = 32 * s + 8 * hh;
            BPack pb;
            pb.q[0] = *reinterpret_cast<const uint4*>(hrow + kb);
            pb.q[1] = *reinterpret_cast<const uint4*>(hrow + kb + 16);
            a2[s] = pb.v;
        }

        // Hoist per-row epilogue state: gather-row pointers + cutoff scalars.
        const float* jrow[8];
        float cut[8];
#pragma unroll
        for (int i = 0; i < 8; ++i) {
            jrow[i] = vproj + (size_t)sJ[wave * 16 + 8 * hh + i] * 128 + l16;
            cut[i]  = HAS_CUT ? sC[wave * 16 + 8 * hh + i] : 1.f;
        }
        float* obase = out + (size_t)(tile * 128 + wave * 16 + 8 * hh) * 128 + l16;

        // GEMM2 + epilogue: W = h@W2^T + b2, *C, * gathered v_[j], store.
#pragma unroll
        for (int t = 0; t < 8; ++t) {
            v8f acc = {};
#pragma unroll
            for (int s = 0; s < 4; ++s) {
                v16bf bm = lds_read_op(&sB2[((t * 4 + s) * 32 + lane) * 8]);
                acc = __builtin_amdgcn_wmma_f32_16x16x32_bf16(
                        false, a2[s], false, bm, (short)0, acc, false, false);
            }
            float bias = sb2[16 * t + l16];
            if (full) {
#pragma unroll
                for (int i = 0; i < 8; ++i) {
                    float wv = (acc[i] + bias) * cut[i];
                    obase[i * 128 + 16 * t] = wv * jrow[i][16 * t];
                }
            } else {
#pragma unroll
                for (int i = 0; i < 8; ++i) {
                    if (tile * 128 + wave * 16 + 8 * hh + i < E) {
                        float wv = (acc[i] + bias) * cut[i];
                        obase[i * 128 + 16 * t] = wv * jrow[i][16 * t];
                    }
                }
            }
        }
    }
}

// ---- host launch ---------------------------------------------------------

extern "C" void kernel_launch(void* const* d_in, const int* in_sizes, int n_in,
                              void* d_out, int out_size, void* d_ws, size_t ws_size,
                              hipStream_t stream)
{
    const float* v        = (const float*)d_in[0];
    const float* dist     = (const float*)d_in[1];
    const float* dist_emb = (const float*)d_in[2];
    const float* fea_hull = (const float*)d_in[3];
    const int*   eidx     = (const int*)d_in[4];   // [2,E]; row 0 = j
    const int*   eidx_h   = (const int*)d_in[5];   // [2,Eh]; row 0 = j_
    const float* lin_w    = (const float*)d_in[6];
    const float* mlp1_w   = (const float*)d_in[7];
    const float* mlp1_b   = (const float*)d_in[8];
    const float* mlp2_w   = (const float*)d_in[9];
    const float* mlp2_b   = (const float*)d_in[10];
    const float* lin_hw   = (const float*)d_in[11];
    const float* mlph1_w  = (const float*)d_in[12];
    const float* mlph1_b  = (const float*)d_in[13];
    const float* mlph2_w  = (const float*)d_in[14];
    const float* mlph2_b  = (const float*)d_in[15];

    const int N  = in_sizes[0] / 128;
    const int E  = in_sizes[1];
    const int Eh = in_sizes[3] / 16;

    float* vproj  = (float*)d_ws;                 // [N,128]
    float* vprojh = vproj + (size_t)N * 128;      // [N,128]
    float* e_out  = (float*)d_out;                // [E,128]
    float* eh_out = e_out + (size_t)E * 128;      // [Eh,128]

    dim3 blk(256);
    int projBlocks = (N + 127) / 128;
    hipLaunchKernelGGL(proj_kernel, dim3(projBlocks), blk, 0, stream,
                       v, lin_w, lin_hw, vproj, vprojh, N);

    int tiles  = (E + 127) / 128;
    int tilesH = (Eh + 127) / 128;
    int g1 = tiles  < 1024 ? tiles  : 1024;
    int g2 = tilesH < 1024 ? tilesH : 1024;
    hipLaunchKernelGGL((edge_kernel<50, 2, true>), dim3(g1), blk, 0, stream,
                       dist_emb, dist, eidx, vproj,
                       mlp1_w, mlp1_b, mlp2_w, mlp2_b, e_out, E, tiles);
    hipLaunchKernelGGL((edge_kernel<16, 1, false>), dim3(g2), blk, 0, stream,
                       fea_hull, nullptr, eidx_h, vprojh,
                       mlph1_w, mlph1_b, mlph2_w, mlph2_b, eh_out, Eh, tilesH);
}
